// Mamba2MacOS_42769284333961
// MI455X (gfx1250) — matrix operand order repaired and checked by reading, not compiled, not run on hardware
//
#include <hip/hip_runtime.h>
#include <hip/hip_bf16.h>

// ---------------------------------------------------------------- types ----
typedef __attribute__((ext_vector_type(16))) __bf16 v16bf;
typedef __attribute__((ext_vector_type(8)))  __bf16 v8bf;
typedef __attribute__((ext_vector_type(8)))  float  v8f;
typedef int b128v __attribute__((vector_size(16)));   // matches builtin param

#define AS1 __attribute__((address_space(1)))
#define AS3 __attribute__((address_space(3)))

#if defined(__has_builtin)
#  if __has_builtin(__builtin_amdgcn_global_load_async_to_lds_b128)
#    define HAVE_ASYNC_LDS 1
#  endif
#  if __has_builtin(__builtin_amdgcn_s_wait_asynccnt)
#    define HAVE_WAIT_ASYNC 1
#  endif
#endif
#ifndef HAVE_ASYNC_LDS
#  define HAVE_ASYNC_LDS 0
#endif
#ifndef HAVE_WAIT_ASYNC
#  define HAVE_WAIT_ASYNC 0
#endif

__device__ __forceinline__ void async_wait_all() {
#if HAVE_ASYNC_LDS
#  if HAVE_WAIT_ASYNC
  __builtin_amdgcn_s_wait_asynccnt(0);
#  else
  asm volatile("s_wait_asynccnt 0x0" ::: "memory");
#  endif
#endif
}

#define D_MODEL   1024
#define D_STATE   64
#define HEADDIM   128
#define D_INNER   2048
#define NHEADS    16
#define SEQ       1024
#define NPROJ     144          // NHEADS + 2*D_STATE
#define NPROJ_PAD 192          // padded to multiple of 64 for the WMMA tiler

__device__ __forceinline__ unsigned short f2bf(float f) {
  unsigned int u = __float_as_uint(f);
  unsigned int r = u + 0x7fffu + ((u >> 16) & 1u);   // round-to-nearest-even
  return (unsigned short)(r >> 16);
}

// ------------------------------------------------- f32 -> bf16 (w/ pad) ----
__global__ void cvt_pad_kernel(const float* __restrict__ src,
                               unsigned short* __restrict__ dst,
                               int src_rows, int dst_rows, int cols) {
  long idx = (long)blockIdx.x * blockDim.x + threadIdx.x;
  long total = (long)dst_rows * cols;
  if (idx >= total) return;
  int row = (int)(idx / cols);
  float v = (row < src_rows) ? src[idx] : 0.0f;
  dst[idx] = f2bf(v);
}

// ------------------------------------------------------- WMMA bf16 GEMM ----
// C[m,n] = sum_k A[m,k] * B[n,k];  A: MxK bf16 row-major, B: NxK bf16 row-major.
// Block = 128 threads (4 waves). Wave computes 32(M) x 64(N); block = 128x64.
// B tile (64n x 32k) is double-buffered in LDS via async load-to-LDS.
__global__ void __launch_bounds__(128)
gemm_bf16_wmma(const unsigned short* __restrict__ A,
               const unsigned short* __restrict__ B,
               float* __restrict__ C,
               int M, int N, int K) {
  __shared__ unsigned short sBuf[2][64 * 32];     // 2 x 4KB

  const int tid  = threadIdx.x;
  const int lane = tid & 31;
  const int wave = tid >> 5;
  const int m0   = blockIdx.y * 128 + wave * 32;
  const int n0   = blockIdx.x * 64;
  const int half = lane >> 4;        // 0 or 1
  const int l16  = lane & 15;

  // stage B[n0..n0+63, k0..k0+31] -> sBuf[bsel]; 256 16-byte chunks, 2/thread
  auto stage = [&](int k0, int bsel) {
#pragma unroll
    for (int u = 0; u < 2; ++u) {
      const int c  = tid + u * 128;
      const int n  = c >> 2;
      const int cw = c & 3;
      const unsigned short* g = B + (size_t)(n0 + n) * K + k0 + cw * 8;
      unsigned short*       l = &sBuf[bsel][n * 32 + cw * 8];
#if HAVE_ASYNC_LDS
      unsigned short* gnc = const_cast<unsigned short*>(g);
      __builtin_amdgcn_global_load_async_to_lds_b128(
          (AS1 b128v*)gnc, (AS3 b128v*)l, 0, 0);
#else
      *reinterpret_cast<v8bf*>(l) = *reinterpret_cast<const v8bf*>(g);
#endif
    }
  };

  v8f acc[2][4] = {};

  const unsigned short* Arow0 = A + (size_t)(m0 + l16) * K;
  const unsigned short* Arow1 = Arow0 + (size_t)16 * K;

  stage(0, 0);
  int bsel = 0;

  for (int k0 = 0; k0 < K; k0 += 32) {
    async_wait_all();
    __syncthreads();
    if (k0 + 32 < K) stage(k0 + 32, bsel ^ 1);

    // A fragments: lanes 0-15 hold K {0..7,16..23}, lanes 16-31 K {8..15,24..31}
    const int akb = k0 + half * 8;
    union { v16bf v; v8bf h[2]; } af0, af1;
    af0.h[0] = *reinterpret_cast<const v8bf*>(Arow0 + akb);
    af0.h[1] = *reinterpret_cast<const v8bf*>(Arow0 + akb + 16);
    af1.h[0] = *reinterpret_cast<const v8bf*>(Arow1 + akb);
    af1.h[1] = *reinterpret_cast<const v8bf*>(Arow1 + akb + 16);

#pragma unroll
    for (int j = 0; j < 4; ++j) {
      // B fragment from LDS: lane = n, lanes 0-15 K {0..7,16..23}, lanes 16-31 K {8..15,24..31}
      const unsigned short* bp = &sBuf[bsel][(j * 16 + l16) * 32 + half * 8];
      union { v16bf v; v8bf h[2]; } bfrag;
      bfrag.h[0] = *reinterpret_cast<const v8bf*>(bp);
      bfrag.h[1] = *reinterpret_cast<const v8bf*>(bp + 16);
      acc[0][j] = __builtin_amdgcn_wmma_f32_16x16x32_bf16(
          false, af0.v, false, bfrag.v, (short)0, acc[0][j], false, false);
      acc[1][j] = __builtin_amdgcn_wmma_f32_16x16x32_bf16(
          false, af1.v, false, bfrag.v, (short)0, acc[1][j], false, false);
    }
    bsel ^= 1;
  }

#pragma unroll
  for (int f = 0; f < 2; ++f) {
#pragma unroll
    for (int j = 0; j < 4; ++j) {
      const int n = n0 + j * 16 + l16;
#pragma unroll
      for (int v = 0; v < 8; ++v) {
        const int m = m0 + f * 16 + v + half * 8;  // D: VGPRv -> M=v (+8 hi half)
        C[(long)m * N + n] = acc[f][j][v];
      }
    }
  }
}

// -------------------------------------- depthwise causal conv4 + SiLU ------
__global__ void conv_silu_kernel(const float* __restrict__ xz,
                                 const float* __restrict__ Wc,
                                 const float* __restrict__ bc,
                                 float* __restrict__ xconv,
                                 unsigned short* __restrict__ xbf) {
  int idx = blockIdx.x * blockDim.x + threadIdx.x;
  if (idx >= SEQ * D_INNER) return;
  int t = idx / D_INNER, c = idx % D_INNER;
  float s = bc[c];
#pragma unroll
  for (int j = 0; j < 4; ++j) {
    int tt = t - 3 + j;
    if (tt >= 0) s += xz[(long)tt * (2 * D_INNER) + c] * Wc[c * 4 + j];
  }
  float v = s / (1.0f + __expf(-s));             // SiLU
  xconv[idx] = v;
  xbf[idx]   = f2bf(v);
}

// --------------------------------- split dt/B/C, bias + softplus(dt) -------
__global__ void split_dtbc_kernel(const float* __restrict__ xp,
                                  const float* __restrict__ b_dt,
                                  const float* __restrict__ dt_bias,
                                  float* __restrict__ dtv,
                                  float* __restrict__ Bm,
                                  float* __restrict__ Cm) {
  int idx = blockIdx.x * blockDim.x + threadIdx.x;
  if (idx >= SEQ * NPROJ) return;
  int t = idx / NPROJ, j = idx % NPROJ;
  float v = xp[(long)t * NPROJ_PAD + j] + b_dt[j];
  if (j < NHEADS) {
    float x = v + dt_bias[j];
    dtv[t * NHEADS + j] = (x > 20.0f) ? x : log1pf(__expf(x));
  } else if (j < NHEADS + D_STATE) {
    Bm[t * D_STATE + (j - NHEADS)] = v;
  } else {
    Cm[t * D_STATE + (j - NHEADS - D_STATE)] = v;
  }
}

// ------------------------------------------ sequential selective scan ------
// One block per head; state[128p x 64n] lives in registers (32 f32 / thread).
__global__ void __launch_bounds__(256)
scan_kernel(const float* __restrict__ A_log, const float* __restrict__ Dp,
            const float* __restrict__ dtv, const float* __restrict__ Bm,
            const float* __restrict__ Cm, const float* __restrict__ xconv,
            float* __restrict__ y) {
  __shared__ float sB[D_STATE], sC[D_STATE], sX[HEADDIM], sPart[256];
  const int h   = blockIdx.x;
  const int tid = threadIdx.x;
  const int p   = tid & (HEADDIM - 1);
  const int nb  = (tid >> 7) * 32;               // this thread's n-chunk
  const int ch  = h * HEADDIM + p;

  float a[32], st[32];
#pragma unroll
  for (int i = 0; i < 32; ++i) {
    a[i]  = -__expf(A_log[(long)ch * D_STATE + nb + i]);
    st[i] = 0.0f;
  }
  const float dD = Dp[ch];

  for (int t = 0; t < SEQ; ++t) {
    if (tid < 64)        sB[tid]       = Bm[t * D_STATE + tid];
    else if (tid < 128)  sC[tid - 64]  = Cm[t * D_STATE + tid - 64];
    else                 sX[tid - 128] = xconv[(long)t * D_INNER + h * HEADDIM + tid - 128];
    const float dt = dtv[t * NHEADS + h];
    __syncthreads();

    const float dx = dt * sX[p];
    float part = 0.0f;
#pragma unroll
    for (int i = 0; i < 32; ++i) {
      float dA = __expf(dt * a[i]);
      st[i] = st[i] * dA + dx * sB[nb + i];
      part += st[i] * sC[nb + i];
    }
    sPart[tid] = part;
    __syncthreads();

    if (tid < 128)
      y[(long)t * D_INNER + h * HEADDIM + tid] =
          sPart[tid] + sPart[tid + 128] + dD * sX[tid];
    __syncthreads();                              // protect sB/sC/sX for next t
  }
}

// ------------------------------------- LayerNorm + sigmoid(z) gate ---------
__global__ void __launch_bounds__(256)
ln_gate_kernel(const float* __restrict__ yscan, const float* __restrict__ xz,
               const float* __restrict__ gamma, const float* __restrict__ beta,
               unsigned short* __restrict__ ybf) {
  __shared__ float ssum[256], ssq[256];
  const int row = blockIdx.x, tid = threadIdx.x;
  const float* yr = yscan + (long)row * D_INNER;
  float s = 0.0f, q = 0.0f;
#pragma unroll
  for (int i = 0; i < 8; ++i) { float v = yr[tid + i * 256]; s += v; q += v * v; }
  ssum[tid] = s; ssq[tid] = q; __syncthreads();
  for (int stp = 128; stp > 0; stp >>= 1) {
    if (tid < stp) { ssum[tid] += ssum[tid + stp]; ssq[tid] += ssq[tid + stp]; }
    __syncthreads();
  }
  const float mu   = ssum[0] * (1.0f / D_INNER);
  const float var  = ssq[0] * (1.0f / D_INNER) - mu * mu;
  const float rstd = rsqrtf(var + 1e-5f);
#pragma unroll
  for (int i = 0; i < 8; ++i) {
    int c = tid + i * 256;
    float v = (yr[c] - mu) * rstd * gamma[c] + beta[c];
    float z = xz[(long)row * (2 * D_INNER) + D_INNER + c];
    v *= 1.0f / (1.0f + __expf(-z));
    ybf[(long)row * D_INNER + c] = f2bf(v);
  }
}

// ------------------------------------------------------------- launch ------
extern "C" void kernel_launch(void* const* d_in, const int* in_sizes, int n_in,
                              void* d_out, int out_size, void* d_ws, size_t ws_size,
                              hipStream_t stream) {
  const float* hidden  = (const float*)d_in[0];   // (1,1024,1024)
  const float* W_in    = (const float*)d_in[1];   // (4096,1024)
  const float* W_conv  = (const float*)d_in[2];   // (2048,1,4)
  const float* b_conv  = (const float*)d_in[3];   // (2048)
  const float* A_log   = (const float*)d_in[4];   // (2048,64)
  const float* Dparam  = (const float*)d_in[5];   // (2048)
  const float* W_dt    = (const float*)d_in[6];   // (144,2048)
  const float* b_dt    = (const float*)d_in[7];   // (144)
  const float* dt_bias = (const float*)d_in[8];   // (16)
  const float* W_out   = (const float*)d_in[9];   // (1024,2048)
  const float* ln_g    = (const float*)d_in[10];  // (2048)
  const float* ln_b    = (const float*)d_in[11];  // (2048)
  float* out = (float*)d_out;                     // (1,1024,1024)

  // workspace carving (256B aligned)
  char* ws = (char*)d_ws;
  size_t off = 0;
  auto carve = [&](size_t bytes) -> void* {
    void* p = ws + off;
    off = (off + bytes + 255) & ~(size_t)255;
    return p;
  };
  unsigned short* hbf    = (unsigned short*)carve((size_t)SEQ * D_MODEL * 2);
  unsigned short* Winbf  = (unsigned short*)carve((size_t)2 * D_INNER * D_MODEL * 2);
  unsigned short* Wdtbf  = (unsigned short*)carve((size_t)NPROJ_PAD * D_INNER * 2);
  unsigned short* Woutbf = (unsigned short*)carve((size_t)D_MODEL * D_INNER * 2);
  float*          xzbuf  = (float*)carve((size_t)SEQ * 2 * D_INNER * 4);
  float*          xconv  = (float*)carve((size_t)SEQ * D_INNER * 4);
  unsigned short* xbf    = (unsigned short*)carve((size_t)SEQ * D_INNER * 2);
  float*          xp     = (float*)carve((size_t)SEQ * NPROJ_PAD * 4);
  float*          dtv    = (float*)carve((size_t)SEQ * NHEADS * 4);
  float*          Bmb    = (float*)carve((size_t)SEQ * D_STATE * 4);
  float*          Cmb    = (float*)carve((size_t)SEQ * D_STATE * 4);
  float*          yscan  = (float*)carve((size_t)SEQ * D_INNER * 4);
  unsigned short* ybf    = (unsigned short*)carve((size_t)SEQ * D_INNER * 2);
  (void)ws_size; (void)n_in; (void)in_sizes; (void)out_size;

  const int TB = 256;
  auto blocks = [](long n, int tb) { return dim3((unsigned)((n + tb - 1) / tb)); };

  // 1) fp32 -> bf16 conversions (weights + activations)
  cvt_pad_kernel<<<blocks((long)SEQ * D_MODEL, TB), TB, 0, stream>>>(hidden, hbf, SEQ, SEQ, D_MODEL);
  cvt_pad_kernel<<<blocks((long)2 * D_INNER * D_MODEL, TB), TB, 0, stream>>>(W_in, Winbf, 2 * D_INNER, 2 * D_INNER, D_MODEL);
  cvt_pad_kernel<<<blocks((long)NPROJ_PAD * D_INNER, TB), TB, 0, stream>>>(W_dt, Wdtbf, NPROJ, NPROJ_PAD, D_INNER);
  cvt_pad_kernel<<<blocks((long)D_MODEL * D_INNER, TB), TB, 0, stream>>>(W_out, Woutbf, D_MODEL, D_MODEL, D_INNER);

  // 2) in-projection: xz = hidden @ W_in^T  (1024 x 4096 x K=1024)
  gemm_bf16_wmma<<<dim3((2 * D_INNER) / 64, SEQ / 128), 128, 0, stream>>>(
      hbf, Winbf, xzbuf, SEQ, 2 * D_INNER, D_MODEL);

  // 3) depthwise causal conv + SiLU
  conv_silu_kernel<<<blocks((long)SEQ * D_INNER, TB), TB, 0, stream>>>(
      xzbuf, W_conv, b_conv, xconv, xbf);

  // 4) dt/B/C projection: xp = x @ W_dt^T  (1024 x 192pad x K=2048)
  gemm_bf16_wmma<<<dim3(NPROJ_PAD / 64, SEQ / 128), 128, 0, stream>>>(
      xbf, Wdtbf, xp, SEQ, NPROJ_PAD, D_INNER);

  // 5) split + softplus
  split_dtbc_kernel<<<blocks((long)SEQ * NPROJ, TB), TB, 0, stream>>>(
      xp, b_dt, dt_bias, dtv, Bmb, Cmb);

  // 6) sequential selective scan (state in registers, per-head blocks)
  scan_kernel<<<NHEADS, 256, 0, stream>>>(A_log, Dparam, dtv, Bmb, Cmb, xconv, yscan);

  // 7) LayerNorm + sigmoid(z) gate -> bf16
  ln_gate_kernel<<<SEQ, 256, 0, stream>>>(yscan, xzbuf, ln_g, ln_b, ybf);

  // 8) out-projection: out = y @ W_out^T (1024 x 1024 x K=2048)
  gemm_bf16_wmma<<<dim3(D_MODEL / 64, SEQ / 128), 128, 0, stream>>>(
      ybf, Woutbf, out, SEQ, D_MODEL, D_INNER);
}
